// MultiHeadedAttention_40157944218115
// MI455X (gfx1250) — matrix operand order repaired and checked by reading, not compiled, not run on hardware
//
#include <hip/hip_runtime.h>
#include <hip/hip_bf16.h>

// ---------------------------------------------------------------------------
// MultiHeadedAttention (RoPE + sliding window, W=50) for MI455X / gfx1250.
// B=2, S=2048, D_MODEL=1024, H=16, D_K=64.
// bf16 WMMA (v_wmma_f32_16x16x32_bf16) everywhere; projection GEMMs use
// double-buffered async global->LDS copies (ASYNCcnt path) with 128x128
// macro tiles to kill L2 re-read traffic. Attention uses a 128-key window
// per 16-query tile so softmax is single-pass.
// ---------------------------------------------------------------------------

typedef __attribute__((ext_vector_type(16))) __bf16 v16bf;
typedef __attribute__((ext_vector_type(8)))  float  v8f;

union Frag {
  v16bf v;
  unsigned short u[16];
  uint4 q[2];
};

__device__ __forceinline__ unsigned short f2bf(float f) {
  union { float f; unsigned int u; } x; x.f = f;
  unsigned int r = x.u + 0x7FFFu + ((x.u >> 16) & 1u);   // round-to-nearest-even
  return (unsigned short)(r >> 16);
}
__device__ __forceinline__ float bf2f(unsigned short h) {
  union { unsigned int u; float f; } x; x.u = ((unsigned int)h) << 16;
  return x.f;
}

__device__ __forceinline__ v8f wmma_bf16(v16bf a, v16bf b, v8f c) {
  // 8 args: (neg_a, A, neg_b, B, c_mod, C, reuse_a, reuse_b)
  return __builtin_amdgcn_wmma_f32_16x16x32_bf16(false, a, false, b,
                                                 (short)0, c, false, false);
}

// flat shared pointer -> 32-bit LDS byte address (ISA: LDS addr = addr[31:0])
__device__ __forceinline__ unsigned lds_addr32(const void* p) {
  return (unsigned)(unsigned long long)p;
}

// async 16B global -> LDS copy (per-lane), tracked by ASYNCcnt
#define ASYNC_CP_B128(ldsu, gptr)                                          \
  asm volatile("global_load_async_to_lds_b128 %0, %1, off"                 \
               :: "v"(ldsu), "v"(gptr) : "memory")

// ---------------------------------------------------------------- fp32->bf16
__global__ void cvt_kernel(const float* __restrict__ src,
                           unsigned short* __restrict__ dst, int n) {
  for (int i = blockIdx.x * blockDim.x + threadIdx.x; i < n;
       i += gridDim.x * blockDim.x)
    dst[i] = f2bf(src[i]);
}

// ------------------------------------------------------------------- GEMM
// C[4096,1024] = A[4096,1024] @ W[1024,1024]^T + bias   (bf16 in, f32 accum)
// B(k,n) = W[n][k] -> B column n is row n of W (contiguous), so A-tiles and
// B-tiles have identical [rows][32] copy shape.
// MODE 0: bf16 output permuted to head layout [B=2, H=16, S=2048, Dk=64]
// MODE 1: fp32 output, standard [row,1024] layout (final projection)
template <int MODE>
__global__ __launch_bounds__(256)
void wmma_gemm(const unsigned short* __restrict__ A,
               const unsigned short* __restrict__ W,
               const float* __restrict__ bias,
               unsigned short* __restrict__ outBf,
               float* __restrict__ outF) {
  __shared__ unsigned short At[2][128 * 32];   // 8 KB per buffer
  __shared__ unsigned short Bt[2][128 * 32];   // 8 KB per buffer

  const int tid = threadIdx.x;
  const int wave = tid >> 5;
  const int lane = tid & 31;
  const int lo = lane & 15, hi = lane >> 4;
  const int wrow = wave & 3;              // 4 row groups of 32
  const int wcol = wave >> 2;             // 2 col groups of 64
  const int rowg = blockIdx.y * 128;
  const int colg = blockIdx.x * 128;

  // copy plan: tile = 128 rows x 32 elems = 512 x 16B chunks, 256 threads
  // -> each thread copies rows (cr) and (cr+64) at element offset ce.
  const int cr = tid >> 2;                // 0..63
  const int ce = (tid & 3) * 8;           // 0,8,16,24

  const unsigned short* gA0 = A + (size_t)(rowg + cr) * 1024 + ce;
  const unsigned short* gA1 = A + (size_t)(rowg + cr + 64) * 1024 + ce;
  const unsigned short* gB0 = W + (size_t)(colg + cr) * 1024 + ce;
  const unsigned short* gB1 = W + (size_t)(colg + cr + 64) * 1024 + ce;

  unsigned lA0[2], lA1[2], lB0[2], lB1[2];
#pragma unroll
  for (int bfi = 0; bfi < 2; ++bfi) {
    lA0[bfi] = lds_addr32(&At[bfi][cr * 32 + ce]);
    lA1[bfi] = lds_addr32(&At[bfi][(cr + 64) * 32 + ce]);
    lB0[bfi] = lds_addr32(&Bt[bfi][cr * 32 + ce]);
    lB1[bfi] = lds_addr32(&Bt[bfi][(cr + 64) * 32 + ce]);
  }

  // prologue: fill buffer 0 (k0 = 0)
  ASYNC_CP_B128(lA0[0], gA0);
  ASYNC_CP_B128(lA1[0], gA1);
  ASYNC_CP_B128(lB0[0], gB0);
  ASYNC_CP_B128(lB1[0], gB1);

  v8f acc[2][4] = {};

  for (int ks = 0; ks < 32; ++ks) {       // K = 1024, step 32
    const int cur = ks & 1;
    if (ks + 1 < 32) {
      const int nxt = cur ^ 1;
      const int ko = (ks + 1) * 32;
      // safe: buffer nxt was last read in iteration ks-1, which ended with
      // a barrier, so every wave is done with it.
      ASYNC_CP_B128(lA0[nxt], gA0 + ko);
      ASYNC_CP_B128(lA1[nxt], gA1 + ko);
      ASYNC_CP_B128(lB0[nxt], gB0 + ko);
      ASYNC_CP_B128(lB1[nxt], gB1 + ko);
      // async loads complete in order: <=4 outstanding => buffer cur landed
      asm volatile("s_wait_asynccnt 0x4" ::: "memory");
    } else {
      asm volatile("s_wait_asynccnt 0x0" ::: "memory");
    }
    __syncthreads();                      // all waves' cur-buffer data in LDS

    const unsigned short* at = &At[cur][0];
    const unsigned short* bt = &Bt[cur][0];

    Frag af[2];
#pragma unroll
    for (int mt = 0; mt < 2; ++mt) {
      // A frag 16x32: lane<16 row=lo K 0..15; lane>=16 K 16..31
      const unsigned short* ap = at + (wrow * 32 + mt * 16 + lo) * 32 + hi * 16;
      af[mt].q[0] = *(const uint4*)(ap);
      af[mt].q[1] = *(const uint4*)(ap + 8);
    }
#pragma unroll
    for (int nt = 0; nt < 4; ++nt) {
      // B frag 32x16: column N = lo; lane<16 K 0..15, lane>=16 K 16..31
      const unsigned short* bp = bt + (wcol * 64 + nt * 16 + lo) * 32 + hi * 16;
      Frag bfr;
      bfr.q[0] = *(const uint4*)(bp);
      bfr.q[1] = *(const uint4*)(bp + 8);
      acc[0][nt] = wmma_bf16(af[0].v, bfr.v, acc[0][nt]);
      acc[1][nt] = wmma_bf16(af[1].v, bfr.v, acc[1][nt]);
    }
    __syncthreads();                      // reads of cur done before overwrite
  }

#pragma unroll
  for (int mt = 0; mt < 2; ++mt)
#pragma unroll
    for (int nt = 0; nt < 4; ++nt) {
      int col = colg + wcol * 64 + nt * 16 + lo;
      float bv = bias[col];
#pragma unroll
      for (int r = 0; r < 8; ++r) {
        int row = rowg + wrow * 32 + mt * 16 + r + hi * 8;  // C: M=r+8*hi, N=lo
        float v = acc[mt][nt][r] + bv;
        if (MODE == 0) {
          int b = row >> 11, s = row & 2047;    // S = 2048
          int h = col >> 6, d = col & 63;       // Dk = 64
          outBf[(((size_t)b * 16 + h) * 2048 + s) * 64 + d] = f2bf(v);
        } else {
          outF[(size_t)row * 1024 + col] = v;
        }
      }
    }
}

// ------------------------------------------------------------------- RoPE
// In-place on [B,H,S,64] bf16; pair (d, d+32), d in 0..31.
__global__ void rope_kernel(unsigned short* __restrict__ buf) {
  int idx = blockIdx.x * blockDim.x + threadIdx.x;   // 2*16*2048*32 = 2M
  if (idx >= 2 * 16 * 2048 * 32) return;
  int d = idx & 31;
  int s = (idx >> 5) & 2047;
  int bh = idx >> 16;
  float ang = (float)s * __powf(10000.0f, -(float)(2 * d) * (1.0f / 64.0f));
  float c = __cosf(ang), sn = __sinf(ang);
  unsigned short* p = buf + (((size_t)bh * 2048) + s) * 64;
  float x1 = bf2f(p[d]);
  float x2 = bf2f(p[d + 32]);
  p[d]      = f2bf(x1 * c - x2 * sn);
  p[d + 32] = f2bf(x2 * c + x1 * sn);
}

// --------------------------------------------------------------- Attention
// One wave per (b,h, 16-query tile). 128-key window covers |i-j|<=50 fully.
__global__ __launch_bounds__(128)
void attn_kernel(const unsigned short* __restrict__ Q,
                 const unsigned short* __restrict__ K,
                 const unsigned short* __restrict__ V,
                 unsigned short* __restrict__ O) {
  __shared__ unsigned short ldsP[4 * 16 * 128];   // 4 KB / wave
  const int wave = threadIdx.x >> 5;
  const int lane = threadIdx.x & 31;
  const int lo = lane & 15, hi = lane >> 4;

  int tile = blockIdx.x * 4 + wave;   // 0..4095
  int qt = tile & 127;
  int bh = tile >> 7;                 // b*16 + h
  int b = bh >> 4, h = bh & 15;

  const unsigned short* Qh = Q + (size_t)bh * 2048 * 64;
  const unsigned short* Kh = K + (size_t)bh * 2048 * 64;
  const unsigned short* Vh = V + (size_t)bh * 2048 * 64;

  const int qrow = qt * 16;
  const int j0 = qrow - 56;           // 128 keys: covers [qrow-50, qrow+65]

  // Q A-fragments (d = 0..31 and 32..63)
  Frag aq0, aq1;
  {
    const unsigned short* qp = Qh + (size_t)(qrow + lo) * 64 + hi * 16;
    aq0.q[0] = *(const uint4*)(qp);
    aq0.q[1] = *(const uint4*)(qp + 8);
    aq1.q[0] = *(const uint4*)(qp + 32);
    aq1.q[1] = *(const uint4*)(qp + 40);
  }

  // scores: 8 key blocks of 16; B column j = key row of K (contiguous)
  v8f sc[8];
#pragma unroll
  for (int jb = 0; jb < 8; ++jb) {
    int j = j0 + jb * 16 + lo;
    int jc = j < 0 ? 0 : (j > 2047 ? 2047 : j);   // clamp; masked later
    const unsigned short* kp = Kh + (size_t)jc * 64 + hi * 16;
    Frag b0, b1;
    b0.q[0] = *(const uint4*)(kp);
    b0.q[1] = *(const uint4*)(kp + 8);
    b1.q[0] = *(const uint4*)(kp + 32);
    b1.q[1] = *(const uint4*)(kp + 40);
    v8f z = {};
    z = wmma_bf16(aq0.v, b0.v, z);
    z = wmma_bf16(aq1.v, b1.v, z);
    sc[jb] = z;
  }

  // mask + scale; lane holds rows i = qrow + r + 8*hi, col j = j0 + jb*16 + lo
  float mx[8], sm[8];
#pragma unroll
  for (int r = 0; r < 8; ++r) mx[r] = -3.0e38f;
#pragma unroll
  for (int jb = 0; jb < 8; ++jb) {
    int j = j0 + jb * 16 + lo;
#pragma unroll
    for (int r = 0; r < 8; ++r) {
      int i = qrow + r + hi * 8;
      int dd = i - j; if (dd < 0) dd = -dd;
      bool ok = (j >= 0) && (j < 2048) && (dd <= 50);
      float v = ok ? sc[jb][r] * 0.125f : -3.0e38f;   // 1/sqrt(64)
      sc[jb][r] = v;
      mx[r] = fmaxf(mx[r], v);
    }
  }
  // row lives in 16 lanes sharing the same hi bit: xor-reduce over bits 0..3
#pragma unroll
  for (int off = 1; off < 16; off <<= 1)
#pragma unroll
    for (int r = 0; r < 8; ++r)
      mx[r] = fmaxf(mx[r], __shfl_xor(mx[r], off, 32));

#pragma unroll
  for (int r = 0; r < 8; ++r) sm[r] = 0.0f;
#pragma unroll
  for (int jb = 0; jb < 8; ++jb)
#pragma unroll
    for (int r = 0; r < 8; ++r) {
      float e = (sc[jb][r] < -1.0e37f) ? 0.0f : __expf(sc[jb][r] - mx[r]);
      sc[jb][r] = e;
      sm[r] += e;
    }
#pragma unroll
  for (int off = 1; off < 16; off <<= 1)
#pragma unroll
    for (int r = 0; r < 8; ++r)
      sm[r] += __shfl_xor(sm[r], off, 32);

  float inv[8];
#pragma unroll
  for (int r = 0; r < 8; ++r) inv[r] = 1.0f / sm[r];   // >=1 (diagonal term)

  // stage P (16x128 bf16) through LDS to re-layout C-frag -> A-frag
  unsigned short* lp = ldsP + wave * (16 * 128);
#pragma unroll
  for (int jb = 0; jb < 8; ++jb)
#pragma unroll
    for (int r = 0; r < 8; ++r)
      lp[(r + hi * 8) * 128 + jb * 16 + lo] = f2bf(sc[jb][r] * inv[r]);
  asm volatile("s_wait_dscnt 0" ::: "memory");

  // out = P @ V  (16x128)@(128x64), 4 K-steps of 32 x 4 output col blocks
  v8f out[4] = {};
#pragma unroll
  for (int k0 = 0; k0 < 128; k0 += 32) {
    Frag ap;   // A frag of P: lane row = lo, K = k0 + hi*16 + {0..15}
    const unsigned short* pp = lp + lo * 128 + k0 + hi * 16;
    ap.q[0] = *(const uint4*)(pp);
    ap.q[1] = *(const uint4*)(pp + 8);
#pragma unroll
    for (int dt = 0; dt < 4; ++dt) {
      Frag bv;   // B frag of V: column d = dt*16+lo, K = k0 + hi*16 + e
#pragma unroll
      for (int e = 0; e < 16; ++e) {
        int j = j0 + k0 + hi * 16 + e;
        int jc = j < 0 ? 0 : (j > 2047 ? 2047 : j);   // P is 0 there anyway
        bv.u[e] = Vh[(size_t)jc * 64 + dt * 16 + lo];
      }
      out[dt] = wmma_bf16(ap.v, bv.v, out[dt]);
    }
  }

  // store into [B, S, 1024] bf16 (feeds final projection GEMM directly)
#pragma unroll
  for (int dt = 0; dt < 4; ++dt)
#pragma unroll
    for (int r = 0; r < 8; ++r) {
      int s = qrow + r + hi * 8;
      int dcol = h * 64 + dt * 16 + lo;
      O[(((size_t)b * 2048) + s) * 1024 + dcol] = f2bf(out[dt][r]);
    }
}

// ---------------------------------------------------------------------------
extern "C" void kernel_launch(void* const* d_in, const int* in_sizes, int n_in,
                              void* d_out, int out_size, void* d_ws,
                              size_t ws_size, hipStream_t stream) {
  (void)in_sizes; (void)n_in; (void)out_size; (void)ws_size;
  const float* query = (const float*)d_in[0];
  const float* key   = (const float*)d_in[1];
  const float* value = (const float*)d_in[2];
  const float* Wq = (const float*)d_in[3];
  const float* bq = (const float*)d_in[4];
  const float* Wk = (const float*)d_in[5];
  const float* bk = (const float*)d_in[6];
  const float* Wv = (const float*)d_in[7];
  const float* bv = (const float*)d_in[8];
  const float* Wo = (const float*)d_in[9];
  const float* bo = (const float*)d_in[10];

  // workspace carve-up (bf16): 3 activations + 4 weights + Q/K/V heads + O
  unsigned short* p = (unsigned short*)d_ws;
  const size_t X = (size_t)4096 * 1024;      // 4M elems
  const size_t Wn = (size_t)1024 * 1024;     // 1M elems
  unsigned short* xq = p; p += X;
  unsigned short* xk = p; p += X;
  unsigned short* xv = p; p += X;
  unsigned short* wqb = p; p += Wn;
  unsigned short* wkb = p; p += Wn;
  unsigned short* wvb = p; p += Wn;
  unsigned short* wob = p; p += Wn;
  unsigned short* Qh = p; p += X;
  unsigned short* Kh = p; p += X;
  unsigned short* Vh = p; p += X;
  unsigned short* Ob = p; p += X;

  cvt_kernel<<<2048, 256, 0, stream>>>(query, xq, (int)X);
  cvt_kernel<<<2048, 256, 0, stream>>>(key,   xk, (int)X);
  cvt_kernel<<<2048, 256, 0, stream>>>(value, xv, (int)X);
  cvt_kernel<<<1024, 256, 0, stream>>>(Wq, wqb, (int)Wn);
  cvt_kernel<<<1024, 256, 0, stream>>>(Wk, wkb, (int)Wn);
  cvt_kernel<<<1024, 256, 0, stream>>>(Wv, wvb, (int)Wn);
  cvt_kernel<<<1024, 256, 0, stream>>>(Wo, wob, (int)Wn);

  dim3 gg(8, 32);   // N/128, M/128
  wmma_gemm<0><<<gg, 256, 0, stream>>>(xq, wqb, bq, Qh, nullptr);
  wmma_gemm<0><<<gg, 256, 0, stream>>>(xk, wkb, bk, Kh, nullptr);
  wmma_gemm<0><<<gg, 256, 0, stream>>>(xv, wvb, bv, Vh, nullptr);

  rope_kernel<<<8192, 256, 0, stream>>>(Qh);
  rope_kernel<<<8192, 256, 0, stream>>>(Kh);

  attn_kernel<<<1024, 128, 0, stream>>>(Qh, Kh, Vh, Ob);

  wmma_gemm<1><<<gg, 256, 0, stream>>>(Ob, wob, bo, nullptr, (float*)d_out);
}
